// Net_SDE_21861383536859
// MI455X (gfx1250) — compile-verified, matrix operand order checked by reading, not run on hardware
//
#include <hip/hip_runtime.h>
#include <hip/hip_bf16.h>
#include <hip/hip_fp16.h>

// ---------------- problem constants ----------------
#define NSTEPS 96
#define MC_PATHS 65536
#define H_STEP (1.0f / 96.0f)
#define RATE 0.025f

// frag image sizes (halfs)
#define FRAG_HALFS_PER_MLP 14336   // L0: 2048 + 3 * 4096
#define TOTAL_FRAG_HALFS   43008   // 3 MLPs

// float-region offsets (in floats) following the half frag image in d_ws
#define F_BIAS 0      // 3 mlps * 4 layers * 64 (layer0 slots unused now)
#define F_WOUT 768    // 3 * 64
#define F_RHO  960    // 96
#define F_ACC  1056   // 4 maturities * 8 strikes * 2 (call,put)
#define F_TOTAL 1120

typedef __attribute__((ext_vector_type(16))) _Float16 v16h;
typedef __attribute__((ext_vector_type(8)))  _Float16 v8h;
typedef __attribute__((ext_vector_type(2)))  _Float16 v2h;
typedef __attribute__((ext_vector_type(8)))  float    v8f;

union U16h { uint4 q[2]; v16h v; };
union U8h  { uint4 q; v8h v; };
union U2h  { unsigned int u; v2h h; };

struct Ptrs { const float* p[34]; };

#if __has_builtin(__builtin_amdgcn_wmma_f16_16x16x32_f16)
#define USE_F16_ACC 1
#else
#define USE_F16_ACC 0
#endif

// ---------------- prep: swizzle weights into WMMA B-fragment layout ----------------
// B-frag layout (mirrors ISA 16-bit A layout, M<->N):
// frag (kchunk c, ntile n): lane L holds col = n*16 + (L&15);
//   vgpr v, half h -> kk = (v>>2)*16 + (L>>4)*8 + (v&3)*2 + h ; k = c*32+kk
// storage: halfIndex = ((c*4+n)*32 + lane)*16 + (v*2+h)  (32B contiguous per lane)
// Layer 0: bias folded in as K-row [Kin] with the input providing a constant 1.0.
__global__ void pack_frags(Ptrs P, _Float16* fragbuf) {
    int job = blockIdx.y;            // 0..11
    int mlp = job >> 2, layer = job & 3;
    int idx = blockIdx.x * 256 + threadIdx.x;
    int count = (layer == 0) ? 2048 : 4096;
    if (idx >= count) return;
    int Wptr = (mlp == 0 ? 2 : (mlp == 1 ? 11 : 20)) + 2 * layer;
    int Kin = (layer == 0) ? ((mlp == 0) ? 3 : 2) : 64;
    const float* W = P.p[Wptr];      // row-major [64][Kin]
    const float* B0 = P.p[Wptr + 1]; // bias (used for layer 0 fold)
    int idx16 = idx & 15;
    int lane  = (idx >> 4) & 31;
    int fn    = idx >> 9;            // c*4 + n
    int n = fn & 3;
    int c = fn >> 2;
    int v = idx16 >> 1, h = idx16 & 1;
    int hi = lane >> 4, col = lane & 15;
    int kk = (v >> 2) * 16 + hi * 8 + (v & 3) * 2 + h;
    int k  = c * 32 + kk;
    int row = n * 16 + col;
    float val;
    if (k < Kin)                         val = W[row * Kin + k];
    else if (layer == 0 && k == Kin)     val = B0[row];          // folded bias row
    else                                 val = 0.0f;
    int base = mlp * FRAG_HALFS_PER_MLP + ((layer == 0) ? 0 : (2048 + (layer - 1) * 4096));
    fragbuf[base + idx] = (_Float16)val;
}

__global__ void pack_misc(Ptrs P, float* gf) {
    int tid = threadIdx.x;           // 1 block, 256 threads
    for (int i = tid; i < 768; i += 256) {        // biases (layer0 entries harmless)
        int mlp = i >> 8, rem = i & 255, layer = rem >> 6, e = rem & 63;
        int bidx = (mlp == 0 ? 3 : (mlp == 1 ? 12 : 21)) + 2 * layer;
        gf[F_BIAS + i] = P.p[bidx][e];
    }
    for (int i = tid; i < 192; i += 256) {        // Wout (1x64 each)
        int mlp = i >> 6, e = i & 63;
        int widx = (mlp == 0 ? 10 : (mlp == 1 ? 19 : 28));
        gf[F_WOUT + i] = P.p[widx][e];
    }
    if (tid < 96) {                               // rho(t) table, tiny MLP, tanh out
        float t = tid * H_STEP;
        const float* W0 = P.p[29]; const float* b0 = P.p[30];
        const float* W1 = P.p[31]; const float* b1 = P.p[32];
        const float* Wo = P.p[33];
        float h1[5], h2[5];
        #pragma unroll
        for (int j = 0; j < 5; ++j) h1[j] = fmaxf(W0[j] * t + b0[j], 0.f);
        #pragma unroll
        for (int i2 = 0; i2 < 5; ++i2) {
            float s2 = b1[i2];
            #pragma unroll
            for (int j = 0; j < 5; ++j) s2 += W1[i2 * 5 + j] * h1[j];
            h2[i2] = fmaxf(s2, 0.f);
        }
        float y = 0.f;
        #pragma unroll
        for (int j = 0; j < 5; ++j) y += Wo[j] * h2[j];
        gf[F_RHO + tid] = tanhf(y);
    }
    if (tid < 64) gf[F_ACC + tid] = 0.f;          // zero payoff accumulators
}

// ---------------- main SDE kernel ----------------
__device__ __forceinline__ v16h ld16(const _Float16* p) {
    U16h u;
    u.q[0] = *(const uint4*)(p);
    u.q[1] = *(const uint4*)(p + 8);
    return u.v;
}

// A-fragment from row-major LDS activations (ISA 16-bit A layout)
__device__ __forceinline__ v16h load_frag_a(const _Float16* act, int ldw, int lane, int ckbase) {
    int m = lane & 15, hi = lane >> 4;
    U16h u;
    u.q[0] = *(const uint4*)(act + m * ldw + ckbase + hi * 8);
    u.q[1] = *(const uint4*)(act + m * ldw + ckbase + 16 + hi * 8);
    return u.v;
}

// packed ReLU on an already-packed f16 pair: single v_pk_max_num_f16
__device__ __forceinline__ v2h relu_pk_u(unsigned int pu) {
    unsigned int r;
    asm("v_pk_max_num_f16 %0, %1, 0" : "=v"(r) : "v"(pu));
    U2h o; o.u = r;
    return o.h;
}

// bias_pk == nullptr -> C starts at zero (layer-0, bias folded into weights)
__device__ __forceinline__ void run_layer(const _Float16* src, int ldw, int kchunks,
                                          const _Float16* frags, const unsigned int* bias_pk,
                                          _Float16* dst, int lane) {
    v16h a0 = load_frag_a(src, ldw, lane, 0);
    v16h a1 = (kchunks == 2) ? load_frag_a(src, ldw, lane, 32) : a0;
    int hi  = lane >> 4;
    int col = lane & 15;
    int m0  = hi * 8;
    #pragma unroll
    for (int nt = 0; nt < 4; ++nt) {
        v16h b0 = ld16(frags + ((0 * 4 + nt) * 32 + lane) * 16);
#if USE_F16_ACC
        // f16 accumulator: D packs rows (2e, 2e+1) per VGPR -> no cvt stage
        U8h c;
        if (bias_pk) {
            unsigned int bp = bias_pk[nt * 16 + col];
            c.q = (uint4){bp, bp, bp, bp};
        } else {
            c.q = (uint4){0u, 0u, 0u, 0u};
        }
        c.v = __builtin_amdgcn_wmma_f16_16x16x32_f16(false, a0, false, b0, (short)0, c.v, false, false);
        if (kchunks == 2) {
            v16h b1 = ld16(frags + ((1 * 4 + nt) * 32 + lane) * 16);
            c.v = __builtin_amdgcn_wmma_f16_16x16x32_f16(false, a1, false, b1, (short)0, c.v, false, false);
        }
        #pragma unroll
        for (int e = 0; e < 4; ++e) {
            v2h p = relu_pk_u(c.q[e]);
            dst[(m0 + 2 * e)     * 64 + nt * 16 + col] = p[0];
            dst[(m0 + 2 * e + 1) * 64 + nt * 16 + col] = p[1];
        }
#else
        // f32 accumulator fallback
        v8f c;
        if (bias_pk) {
            U2h b2; b2.u = bias_pk[nt * 16 + col];
            float bv = (float)b2.h[0];
            c = (v8f){bv, bv, bv, bv, bv, bv, bv, bv};
        } else {
            c = (v8f){0.f, 0.f, 0.f, 0.f, 0.f, 0.f, 0.f, 0.f};
        }
        c = __builtin_amdgcn_wmma_f32_16x16x32_f16(false, a0, false, b0, (short)0, c, false, false);
        if (kchunks == 2) {
            v16h b1 = ld16(frags + ((1 * 4 + nt) * 32 + lane) * 16);
            c = __builtin_amdgcn_wmma_f32_16x16x32_f16(false, a1, false, b1, (short)0, c, false, false);
        }
        #pragma unroll
        for (int v = 0; v < 8; v += 2) {
            U2h u; u.h = (v2h)__builtin_amdgcn_cvt_pkrtz(c[v], c[v + 1]);
            v2h p = relu_pk_u(u.u);
            dst[(m0 + v)     * 64 + nt * 16 + col] = p[0];
            dst[(m0 + v + 1) * 64 + nt * 16 + col] = p[1];
        }
#endif
    }
}

// 1x64 output layer: vector LDS loads + v_fma_mix_f32 (f16 sources, f32 accum)
__device__ __forceinline__ float out_layer(const _Float16* act, const _Float16* wout_h, int lane) {
    int m = lane & 15, hi = lane >> 4;
    const _Float16* arow = act + m * 64 + hi * 32;     // 32 contiguous halfs, 16B aligned
    const _Float16* wrow = wout_h + hi * 32;
    union UV { uint4 q; v2h h[4]; };
    UV ua[4], uw[4];
    #pragma unroll
    for (int i = 0; i < 4; ++i) {
        ua[i].q = *(const uint4*)(arow + i * 8);
        uw[i].q = *(const uint4*)(wrow + i * 8);
    }
    float s = 0.f;
    #pragma unroll
    for (int i = 0; i < 4; ++i) {
        #pragma unroll
        for (int j = 0; j < 4; ++j) {
            s = fmaf((float)ua[i].h[j][0], (float)uw[i].h[j][0], s);  // v_fma_mix_f32
            s = fmaf((float)ua[i].h[j][1], (float)uw[i].h[j][1], s);
        }
    }
    s += __shfl_xor(s, 16);          // combine the two halves
    return s;
}

__device__ __forceinline__ float softplus_f(float x) {
    return (x > 20.f) ? x : log1pf(__expf(x));
}

__global__ void __launch_bounds__(256, 1)
sde_main(const float* __restrict__ dW, const float* __restrict__ dB,
         const _Float16* __restrict__ gfrag, const float* __restrict__ gf,
         float* __restrict__ gacc) {
    __shared__ __align__(16) _Float16     s_frag[TOTAL_FRAG_HALFS];
    __shared__ __align__(16) unsigned int s_bias_pk[768];   // half2(bias,bias) per entry
    __shared__ __align__(16) _Float16     s_wout_h[192];
    __shared__ __align__(16) float        s_rho[96];
    __shared__ __align__(16) _Float16     s_act[8][16 * 64];
    __shared__ __align__(16) _Float16     s_inp[8][16 * 32];

    int tid = threadIdx.x;
    // stage weights into LDS (shared by all 8 waves, reused over 96 steps)
    {
        const uint4* src = (const uint4*)gfrag;
        uint4* dst = (uint4*)s_frag;
        for (int i = tid; i < TOTAL_FRAG_HALFS / 8; i += 256) dst[i] = src[i];
        for (int i = tid; i < 768; i += 256) {
            U2h b2;
            b2.h[0] = (_Float16)gf[F_BIAS + i];
            b2.h[1] = b2.h[0];
            s_bias_pk[i] = b2.u;
        }
        for (int i = tid; i < 192; i += 256) s_wout_h[i] = (_Float16)gf[F_WOUT + i];
        if (tid < 96) s_rho[tid] = gf[F_RHO + tid];
        // zero input pads once; per step we only rewrite cols 0..3
        _Float16* z = &s_inp[0][0];
        for (int i = tid; i < 8 * 16 * 32; i += 256) z[i] = (_Float16)0.f;
    }
    __syncthreads();

    int w    = tid >> 5;
    int lane = tid & 31;
    int m    = lane & 15;
    int path = blockIdx.x * 128 + w * 16 + m;

    _Float16* inp = s_inp[w];
    _Float16* act = s_act[w];

    const float KS[8] = {0.7f, 0.8f, 0.9f, 0.95f, 1.0f, 1.05f, 1.1f, 1.2f};

    float S = 1.0f, V = 0.04f;
    for (int s = 0; s < NSTEPS; ++s) {
        float t   = s * H_STEP;
        float dw  = dW[path * (NSTEPS + 1) + s];
        float db  = dB[path * (NSTEPS + 1) + s];
        float rho = s_rho[s];

        // ---- diffusion MLP: input [t, S, V, 1] (1.0 activates folded bias row) ----
        if (lane < 16) {
            inp[m * 32 + 0] = (_Float16)t;
            inp[m * 32 + 1] = (_Float16)S;
            inp[m * 32 + 2] = (_Float16)V;
            inp[m * 32 + 3] = (_Float16)1.0f;
        }
        {
            const _Float16*     fr = s_frag + 0 * FRAG_HALFS_PER_MLP;
            const unsigned int* bs = s_bias_pk + 0 * 256;
            run_layer(inp, 32, 1, fr,                nullptr,  act, lane);
            run_layer(act, 64, 2, fr + 2048,         bs + 64,  act, lane);
            run_layer(act, 64, 2, fr + 2048 + 4096,  bs + 128, act, lane);
            run_layer(act, 64, 2, fr + 2048 + 8192,  bs + 192, act, lane);
        }
        float sp_diff = softplus_f(out_layer(act, s_wout_h + 0 * 64, lane));

        // ---- V-MLPs: input [t, V, 1, 0] ----
        if (lane < 16) {
            inp[m * 32 + 1] = (_Float16)V;
            inp[m * 32 + 2] = (_Float16)1.0f;
            inp[m * 32 + 3] = (_Float16)0.f;
        }
        // diffusionV (softplus)
        {
            const _Float16*     fr = s_frag + 1 * FRAG_HALFS_PER_MLP;
            const unsigned int* bs = s_bias_pk + 1 * 256;
            run_layer(inp, 32, 1, fr,                nullptr,  act, lane);
            run_layer(act, 64, 2, fr + 2048,         bs + 64,  act, lane);
            run_layer(act, 64, 2, fr + 2048 + 4096,  bs + 128, act, lane);
            run_layer(act, 64, 2, fr + 2048 + 8192,  bs + 192, act, lane);
        }
        float sp_diffV = softplus_f(out_layer(act, s_wout_h + 1 * 64, lane));
        // driftV (no activation)
        {
            const _Float16*     fr = s_frag + 2 * FRAG_HALFS_PER_MLP;
            const unsigned int* bs = s_bias_pk + 2 * 256;
            run_layer(inp, 32, 1, fr,                nullptr,  act, lane);
            run_layer(act, 64, 2, fr + 2048,         bs + 64,  act, lane);
            run_layer(act, 64, 2, fr + 2048 + 4096,  bs + 128, act, lane);
            run_layer(act, 64, 2, fr + 2048 + 8192,  bs + 192, act, lane);
        }
        float y_drift = out_layer(act, s_wout_h + 2 * 64, lane);

        // ---- Euler update ----
        S = fmaxf(S * (1.0f + RATE * H_STEP) + sp_diff * dw, 0.0f);
        float mix = sqrtf(fmaxf(1.0f - rho * rho, 0.0f)) * db + rho * dw;
        V = fmaxf(V + y_drift * H_STEP + sp_diffV * mix, 0.0f);

        // ---- maturities: accumulate payoffs ----
        if (s == 23 || s == 47 || s == 71 || s == 95) {
            int mi = s / 24;
            #pragma unroll
            for (int kx = 0; kx < 8; ++kx) {
                float cp = fmaxf(S - KS[kx], 0.f);
                float pp = fmaxf(KS[kx] - S, 0.f);
                if (lane >= 16) { cp = 0.f; pp = 0.f; }   // halves are replicated
                #pragma unroll
                for (int off = 16; off >= 1; off >>= 1) {
                    cp += __shfl_xor(cp, off);
                    pp += __shfl_xor(pp, off);
                }
                if (lane == 0) {
                    atomicAdd(&gacc[(mi * 8 + kx) * 2 + 0], cp);
                    atomicAdd(&gacc[(mi * 8 + kx) * 2 + 1], pp);
                }
            }
        }
    }
}

// ---------------- finalize: mean * discount, reference output order ----------------
__global__ void finalize(const float* __restrict__ acc, float* __restrict__ out) {
    int o = threadIdx.x;
    if (o >= 64) return;
    int r = o >> 3, kx = o & 7;
    int mi = (r < 4) ? r : (r - 4);
    int cp = (r < 4) ? 0 : 1;
    float tmat = (float)(24 * (mi + 1)) * H_STEP;
    out[o] = acc[(mi * 8 + kx) * 2 + cp] * (1.0f / (float)MC_PATHS) * expf(-RATE * tmat);
}

extern "C" void kernel_launch(void* const* d_in, const int* in_sizes, int n_in,
                              void* d_out, int out_size, void* d_ws, size_t ws_size,
                              hipStream_t stream) {
    Ptrs P;
    for (int i = 0; i < 34; ++i) P.p[i] = (i < n_in) ? (const float*)d_in[i] : nullptr;

    _Float16* fragbuf = (_Float16*)d_ws;
    float* gf = (float*)((char*)d_ws + (size_t)TOTAL_FRAG_HALFS * sizeof(_Float16));

    pack_frags<<<dim3(16, 12), 256, 0, stream>>>(P, fragbuf);
    pack_misc<<<1, 256, 0, stream>>>(P, gf);
    sde_main<<<MC_PATHS / 128, 256, 0, stream>>>(
        (const float*)d_in[0], (const float*)d_in[1], fragbuf, gf, gf + F_ACC);
    finalize<<<1, 64, 0, stream>>>(gf + F_ACC, (float*)d_out);
}
